// NetworkLayer_8684423872950
// MI455X (gfx1250) — matrix operand order repaired and checked by reading, not compiled, not run on hardware
//
#include <hip/hip_runtime.h>
#include <hip/hip_bf16.h>

typedef __attribute__((ext_vector_type(16))) __bf16 v16bf;
typedef __attribute__((ext_vector_type(8)))  float  v8f;

#define NB    8
#define LSEQ  2048
#define DM    128
#define DI    256
#define DS    16
#define DR    8
#define NTOK  (NB * LSEQ)   // 16384
#define CL    32            // scan chunk length
#define NCH   (LSEQ / CL)   // 64 chunks per batch

// ---------------------------------------------------------------------------
// gfx1250 async global->LDS staging (bypasses VGPRs), via inline CDNA5 asm.
// VDST = per-lane LDS byte offset (low 32 bits of generic shared pointer),
// VADDR = 64-bit global address. Tracked by ASYNCcnt.
// ---------------------------------------------------------------------------
#if defined(__gfx1250__)
#define ASYNC_LDS 1
#else
#define ASYNC_LDS 0
#endif

__device__ __forceinline__ void lds_copy16(const float* g, float* l)
{
#if ASYNC_LDS
    asm volatile("global_load_async_to_lds_b128 %0, %1, off"
                 :: "v"((unsigned)(size_t)l), "v"(g)
                 : "memory");
#else
    *(float4*)l = *(const float4*)g;
#endif
}
__device__ __forceinline__ void lds_copy_wait()
{
#if ASYNC_LDS
    asm volatile("s_wait_asynccnt 0x0" ::: "memory");
#endif
}

// ---------------------------------------------------------------------------
// Kernel 1: LayerNorm (16-token tile in LDS) + in_proj GEMM via bf16 WMMA.
//   xz[tok][n] = sum_k norm(x)[tok][k] * in_proj_w[n][k],  K=128, N=512
// ---------------------------------------------------------------------------
__global__ __launch_bounds__(128) void k_ln_inproj(
    const float* __restrict__ x, const float* __restrict__ nw,
    const float* __restrict__ nbias, const float* __restrict__ W,
    float* __restrict__ xz)
{
    __shared__ float ha[16][132];   // row stride 132 -> 4-bank offset per row
    const int mt  = blockIdx.x;     // 1024 M-tiles of 16 tokens
    const int tid = threadIdx.x;

    // async-stage 16x128 input tile (512 x b128 chunks)
    for (int i = tid; i < 16 * 32; i += 128) {
        int r = i >> 5, c4 = (i & 31) << 2;
        lds_copy16(&x[(mt * 16 + r) * DM + c4], &ha[r][c4]);
    }
    lds_copy_wait();
    __syncthreads();

    if (tid < 16) {                 // per-token layernorm
        float s = 0.f, ss = 0.f;
        for (int c = 0; c < DM; ++c) { float v = ha[tid][c]; s += v; ss += v * v; }
        float mu  = s * (1.f / DM);
        float var = ss * (1.f / DM) - mu * mu;
        float rs  = rsqrtf(var + 1e-5f);
        for (int c = 0; c < DM; ++c)
            ha[tid][c] = (ha[tid][c] - mu) * rs * nw[c] + nbias[c];
    }
    __syncthreads();

    const int wave = tid >> 5, lane = tid & 31;
    const int row = lane & 15, khalf = (lane >> 4) * 16;
    const int col = lane & 15, kboff = (lane >> 4) * 16;
    const int mb  = (lane >> 4) * 8;

    for (int nt = wave; nt < 32; nt += 4) {        // uniform across wave
        const int nb0 = nt * 16;
        v8f acc = {};
        #pragma unroll
        for (int kc = 0; kc < DM; kc += 32) {
            v16bf a, bf;
            const float* ar = &ha[row][kc + khalf];
            #pragma unroll
            for (int j = 0; j < 8; ++j) { a[j] = (__bf16)ar[j]; a[8 + j] = (__bf16)ar[8 + j]; }
            const float* wr = &W[(nb0 + col) * DM + kc + kboff];
            #pragma unroll
            for (int j = 0; j < 16; ++j) bf[j] = (__bf16)wr[j];
            acc = __builtin_amdgcn_wmma_f32_16x16x32_bf16(
                false, a, false, bf, (short)0, acc, false, false);
        }
        #pragma unroll
        for (int r = 0; r < 8; ++r)
            xz[(mt * 16 + mb + r) * (2 * DI) + nb0 + col] = acc[r];
    }
}

// ---------------------------------------------------------------------------
// Kernel 2: depthwise causal conv (D_CONV=4) + SiLU.  xm = xz[..., :256]
// ---------------------------------------------------------------------------
__global__ __launch_bounds__(256) void k_conv(
    const float* __restrict__ xz, const float* __restrict__ cw,
    const float* __restrict__ cb, float* __restrict__ xc)
{
    int idx = blockIdx.x * 256 + threadIdx.x;      // over B*L*DI = 2^22
    int c = idx & (DI - 1);
    int t = (idx >> 8) & (LSEQ - 1);
    int b = idx >> 19;
    float acc = cb[c];
    #pragma unroll
    for (int k = 0; k < 4; ++k) {
        int tt = t + k - 3;
        if (tt >= 0) acc += xz[(b * LSEQ + tt) * (2 * DI) + c] * cw[c * 4 + k];
    }
    xc[idx] = acc / (1.f + __expf(-acc));          // silu
}

// ---------------------------------------------------------------------------
// Kernel 3: x_proj (N=40) + dt_proj (K=8) + softplus. One token per block.
// ---------------------------------------------------------------------------
__global__ __launch_bounds__(256) void k_xproj(
    const float* __restrict__ xc, const float* __restrict__ xpw,
    const float* __restrict__ dpw, const float* __restrict__ dpb,
    float* __restrict__ delta, float* __restrict__ Bm, float* __restrict__ Cm)
{
    __shared__ float sx[DI];
    __shared__ float sdt[DR];
    const int tok = blockIdx.x, tid = threadIdx.x;
    sx[tid] = xc[tok * DI + tid];
    __syncthreads();
    if (tid < DR + 2 * DS) {                       // 40 rows of x_proj_w
        float acc = 0.f;
        for (int k = 0; k < DI; ++k) acc += sx[k] * xpw[tid * DI + k];
        if (tid < DR)            sdt[tid]                 = acc;
        else if (tid < DR + DS)  Bm[tok * DS + tid - DR]  = acc;
        else                     Cm[tok * DS + tid - DR - DS] = acc;
    }
    __syncthreads();
    float acc = dpb[tid];
    #pragma unroll
    for (int r = 0; r < DR; ++r) acc += sdt[r] * dpw[tid * DR + r];
    delta[tok * DI + tid] = (acc > 20.f) ? acc : log1pf(__expf(acc));  // softplus
}

// ---------------------------------------------------------------------------
// Chunked parallel scan, pass 1: per chunk compute S (state from zero start)
// and chunk decay P[n] = exp(Av[n] * sum(delta)).  Grid = NB*NCH blocks.
// ---------------------------------------------------------------------------
__global__ __launch_bounds__(256) void k_scan_partial(
    const float* __restrict__ delta, const float* __restrict__ xc,
    const float* __restrict__ Bm, const float* __restrict__ A_log,
    float* __restrict__ Pws, float* __restrict__ Sws)
{
    __shared__ float sB[CL][DS];
    const int b = blockIdx.x / NCH, k = blockIdx.x % NCH;
    const int c = threadIdx.x;
    const int tok0 = b * LSEQ + k * CL;
    for (int i = c; i < CL * DS; i += 256) sB[i / DS][i % DS] = Bm[tok0 * DS + i];
    __syncthreads();

    float Av[DS], h[DS];
    #pragma unroll
    for (int n = 0; n < DS; ++n) { Av[n] = -__expf(A_log[c * DS + n]); h[n] = 0.f; }
    float dsum = 0.f;
    for (int t = 0; t < CL; ++t) {
        const int tok = tok0 + t;
        float d  = delta[tok * DI + c];
        float dx = d * xc[tok * DI + c];
        dsum += d;
        #pragma unroll
        for (int n = 0; n < DS; ++n)
            h[n] = __expf(d * Av[n]) * h[n] + dx * sB[t][n];
    }
    const size_t base = ((size_t)(b * NCH + k) * DI + c) * DS;
    #pragma unroll
    for (int n = 0; n < DS; ++n) {
        Pws[base + n] = __expf(dsum * Av[n]);   // prod of per-step decays
        Sws[base + n] = h[n];
    }
}

// Pass 2: serial combine over 64 chunk summaries -> per-chunk start states.
__global__ __launch_bounds__(256) void k_scan_combine(
    const float* __restrict__ Pws, const float* __restrict__ Sws,
    float* __restrict__ Hs)
{
    const int b = blockIdx.x, c = threadIdx.x;
    float h[DS];
    #pragma unroll
    for (int n = 0; n < DS; ++n) h[n] = 0.f;
    for (int k = 0; k < NCH; ++k) {
        const size_t base = ((size_t)(b * NCH + k) * DI + c) * DS;
        #pragma unroll
        for (int n = 0; n < DS; ++n) {
            Hs[base + n] = h[n];
            h[n] = Pws[base + n] * h[n] + Sws[base + n];
        }
    }
}

// Pass 3: replay each chunk from its start state; emit gated output y.
__global__ __launch_bounds__(256) void k_scan_apply(
    const float* __restrict__ xz, const float* __restrict__ xc,
    const float* __restrict__ delta, const float* __restrict__ Bm,
    const float* __restrict__ Cm, const float* __restrict__ A_log,
    const float* __restrict__ Dp, const float* __restrict__ Hs,
    float* __restrict__ y)
{
    __shared__ float sB[CL][DS], sC[CL][DS];
    const int b = blockIdx.x / NCH, k = blockIdx.x % NCH;
    const int c = threadIdx.x;
    const int tok0 = b * LSEQ + k * CL;
    for (int i = c; i < CL * DS; i += 256) {
        sB[i / DS][i % DS] = Bm[tok0 * DS + i];
        sC[i / DS][i % DS] = Cm[tok0 * DS + i];
    }
    __syncthreads();

    const size_t base = ((size_t)(b * NCH + k) * DI + c) * DS;
    float Av[DS], h[DS];
    #pragma unroll
    for (int n = 0; n < DS; ++n) {
        Av[n] = -__expf(A_log[c * DS + n]);
        h[n]  = Hs[base + n];
    }
    const float Dv = Dp[c];
    for (int t = 0; t < CL; ++t) {
        const int tok = tok0 + t;
        float d  = delta[tok * DI + c];
        float xv = xc[tok * DI + c];
        float dx = d * xv, yv = 0.f;
        #pragma unroll
        for (int n = 0; n < DS; ++n) {
            h[n] = __expf(d * Av[n]) * h[n] + dx * sB[t][n];
            yv  += h[n] * sC[t][n];
        }
        yv += xv * Dv;
        float z = xz[tok * (2 * DI) + DI + c];
        yv *= z / (1.f + __expf(-z));
        y[tok * DI + c] = yv;
    }
}

// Fallback: fully sequential scan (used only if workspace is too small).
__global__ __launch_bounds__(256) void k_scan_seq(
    const float* __restrict__ xz, const float* __restrict__ xc,
    const float* __restrict__ delta, const float* __restrict__ Bm,
    const float* __restrict__ Cm, const float* __restrict__ A_log,
    const float* __restrict__ Dp, float* __restrict__ y)
{
    __shared__ float sB[DS], sC[DS];
    const int b = blockIdx.x, c = threadIdx.x;
    float Av[DS], h[DS];
    #pragma unroll
    for (int n = 0; n < DS; ++n) { Av[n] = -__expf(A_log[c * DS + n]); h[n] = 0.f; }
    const float Dv = Dp[c];
    for (int t = 0; t < LSEQ; ++t) {
        const int tok = b * LSEQ + t;
        if (c < DS)          sB[c]      = Bm[tok * DS + c];
        else if (c < 2 * DS) sC[c - DS] = Cm[tok * DS + c - DS];
        __syncthreads();
        float d = delta[tok * DI + c], xv = xc[tok * DI + c];
        float dx = d * xv, yv = 0.f;
        #pragma unroll
        for (int n = 0; n < DS; ++n) {
            h[n] = __expf(d * Av[n]) * h[n] + dx * sB[n];
            yv  += h[n] * sC[n];
        }
        yv += xv * Dv;
        float z = xz[tok * (2 * DI) + DI + c];
        yv *= z / (1.f + __expf(-z));
        y[tok * DI + c] = yv;
        __syncthreads();
    }
}

// ---------------------------------------------------------------------------
// Kernel 5: out_proj GEMM via bf16 WMMA.  K=256, N=128.
// ---------------------------------------------------------------------------
__global__ __launch_bounds__(128) void k_outproj(
    const float* __restrict__ y, const float* __restrict__ W,
    float* __restrict__ out)
{
    __shared__ float ha[16][260];
    const int mt = blockIdx.x, tid = threadIdx.x;
    for (int i = tid; i < 16 * 64; i += 128) {     // 1024 b128 chunks
        int r = i >> 6, c4 = (i & 63) << 2;
        lds_copy16(&y[(mt * 16 + r) * DI + c4], &ha[r][c4]);
    }
    lds_copy_wait();
    __syncthreads();

    const int wave = tid >> 5, lane = tid & 31;
    const int row = lane & 15, khalf = (lane >> 4) * 16;
    const int col = lane & 15, kboff = (lane >> 4) * 16;
    const int mb  = (lane >> 4) * 8;

    for (int nt = wave; nt < 8; nt += 4) {
        const int nb0 = nt * 16;
        v8f acc = {};
        #pragma unroll
        for (int kc = 0; kc < DI; kc += 32) {
            v16bf a, bf;
            const float* ar = &ha[row][kc + khalf];
            #pragma unroll
            for (int j = 0; j < 8; ++j) { a[j] = (__bf16)ar[j]; a[8 + j] = (__bf16)ar[8 + j]; }
            const float* wr = &W[(nb0 + col) * DI + kc + kboff];
            #pragma unroll
            for (int j = 0; j < 16; ++j) bf[j] = (__bf16)wr[j];
            acc = __builtin_amdgcn_wmma_f32_16x16x32_bf16(
                false, a, false, bf, (short)0, acc, false, false);
        }
        #pragma unroll
        for (int r = 0; r < 8; ++r)
            out[(mt * 16 + mb + r) * DM + nb0 + col] = acc[r];
    }
}

__global__ __launch_bounds__(256) void k_copy(
    const float* __restrict__ src, float* __restrict__ dst, int n)
{
    int i = blockIdx.x * 256 + threadIdx.x;
    if (i < n) dst[i] = src[i];
}

extern "C" void kernel_launch(void* const* d_in, const int* in_sizes, int n_in,
                              void* d_out, int out_size, void* d_ws, size_t ws_size,
                              hipStream_t stream)
{
    (void)in_sizes; (void)n_in; (void)out_size;
    const float* x    = (const float*)d_in[0];
    const float* nw   = (const float*)d_in[1];
    const float* nb   = (const float*)d_in[2];
    const float* ipw  = (const float*)d_in[3];
    const float* cw   = (const float*)d_in[4];
    const float* cb   = (const float*)d_in[5];
    const float* xpw  = (const float*)d_in[6];
    const float* dpw  = (const float*)d_in[7];
    const float* dpb  = (const float*)d_in[8];
    const float* Alog = (const float*)d_in[9];
    const float* Dp   = (const float*)d_in[10];
    const float* opw  = (const float*)d_in[11];

    float* ws    = (float*)d_ws;
    float* xz    = ws;                                   // 16384*512
    float* xc    = xz + (size_t)NTOK * 2 * DI;           // 16384*256
    float* delta = xc + (size_t)NTOK * DI;               // 16384*256
    float* Bm    = delta + (size_t)NTOK * DI;            // 16384*16
    float* Cm    = Bm + (size_t)NTOK * DS;               // 16384*16
    float* yb    = Cm + (size_t)NTOK * DS;               // 16384*256
    const size_t CHNK = (size_t)NB * NCH * DI * DS;      // 2,097,152 floats
    float* Pws   = yb + (size_t)NTOK * DI;
    float* Sws   = Pws + CHNK;
    float* Hs    = Sws + CHNK;
    const size_t need_chunked = (size_t)(Hs + CHNK - ws) * sizeof(float);

    float* out = (float*)d_out;                          // (B,L,128)
    float* res = out + (size_t)NTOK * DM;                // residual copy

    k_ln_inproj<<<NTOK / 16, 128, 0, stream>>>(x, nw, nb, ipw, xz);
    k_conv<<<(NTOK * DI) / 256, 256, 0, stream>>>(xz, cw, cb, xc);
    k_xproj<<<NTOK, 256, 0, stream>>>(xc, xpw, dpw, dpb, delta, Bm, Cm);

    if (ws_size >= need_chunked) {
        k_scan_partial<<<NB * NCH, 256, 0, stream>>>(delta, xc, Bm, Alog, Pws, Sws);
        k_scan_combine<<<NB, 256, 0, stream>>>(Pws, Sws, Hs);
        k_scan_apply<<<NB * NCH, 256, 0, stream>>>(xz, xc, delta, Bm, Cm, Alog, Dp, Hs, yb);
    } else {
        k_scan_seq<<<NB, 256, 0, stream>>>(xz, xc, delta, Bm, Cm, Alog, Dp, yb);
    }

    k_outproj<<<NTOK / 16, 128, 0, stream>>>(yb, opw, out);
    k_copy<<<(NTOK * DM + 255) / 256, 256, 0, stream>>>(x, res, NTOK * DM);
}
